// ConsineMultiHeadAttention_29764123361768
// MI455X (gfx1250) — matrix operand order repaired and checked by reading, not compiled, not run on hardware
//
#include <hip/hip_runtime.h>

// ---------------------------------------------------------------------------
// Cosine multi-head attention (h=1): B=8, S=4096, D_MODEL=512, H_DIM=120
//   ctx   = (query @ w_q^T + b_q) * weight_tensor        [B,S,120]
//   ctx_n = L2-normalize(ctx, axis=-1)
//   out   = softmax(mask ? ctx_n @ ctx_n^T : -1e9, -1)   [B,S,S]
// Plan:
//   K1: projection + scale + L2 norm -> bf16 features [B*S, 128] in d_ws
//   K2: per (batch, 16-row tile): scores via v_wmma_f32_16x16x32_bf16
//       (double-buffered B operands) into a 257 KB LDS slab, then in-LDS
//       masked softmax with float4/int4 vector traffic, write probs.
// ---------------------------------------------------------------------------

typedef __attribute__((ext_vector_type(16))) __bf16 v16bf;
typedef __attribute__((ext_vector_type(8)))  __bf16 v8bf;
typedef __attribute__((ext_vector_type(8)))  float  v8f;

#define B_    8
#define S_    4096
#define DM    512
#define HD    120
#define HDP   128          // K padded to 128 (4 x 32-K WMMA chunks)
#define PITCH 4104         // 4096 + 8 floats: stagger LDS banks across rows
#define NTILE (S_ / 16)    // 256 column tiles

// ------------------------- Kernel 1: project + normalize -------------------
// 128 threads / block, 8 query rows / block. Thread k (<120) computes the
// k-th feature for all 8 rows, reusing each w_q element 8x from registers.
__global__ void __launch_bounds__(128)
proj_norm_kernel(const float* __restrict__ q,
                 const float* __restrict__ wq,
                 const float* __restrict__ bq,
                 const float* __restrict__ wt,
                 __bf16* __restrict__ ctx)
{
    __shared__ float qs[8][DM];     // 16 KB query slab
    __shared__ float norms[8];

    const int  tid     = threadIdx.x;
    const long rowBase = (long)blockIdx.x * 8;

    // cooperative float4 load of 8 query rows (1024 float4 / 128 threads)
    {
        const float4* q4 = (const float4*)(q + rowBase * DM);
        float4*       s4 = (float4*)&qs[0][0];
        for (int i = tid; i < 8 * DM / 4; i += 128) s4[i] = q4[i];
    }
    if (tid < 8) norms[tid] = 0.0f;
    __syncthreads();

    float acc[8];
#pragma unroll
    for (int r = 0; r < 8; ++r) acc[r] = 0.0f;

    const int k = tid;              // feature index, active for k < 120
    if (k < HD) {
        const float4* w4 = (const float4*)(wq + (long)k * DM);
        for (int d4 = 0; d4 < DM / 4; ++d4) {
            const float4 w = w4[d4];
            const int d = d4 * 4;
#pragma unroll
            for (int r = 0; r < 8; ++r) {
                acc[r] = fmaf(qs[r][d + 0], w.x, acc[r]);
                acc[r] = fmaf(qs[r][d + 1], w.y, acc[r]);
                acc[r] = fmaf(qs[r][d + 2], w.z, acc[r]);
                acc[r] = fmaf(qs[r][d + 3], w.w, acc[r]);
            }
        }
        const float bias = bq[k];
        const float scl  = wt[k];   // weight_tensor[0][k]
#pragma unroll
        for (int r = 0; r < 8; ++r) {
            acc[r] = (acc[r] + bias) * scl;
            atomicAdd(&norms[r], acc[r] * acc[r]);   // ds_add_f32
        }
    }
    __syncthreads();

#pragma unroll
    for (int r = 0; r < 8; ++r) {
        const float inv = 1.0f / fmaxf(sqrtf(norms[r]), 1e-12f);
        __bf16 v = (__bf16)0.0f;
        if (k < HD) v = (__bf16)(acc[r] * inv);
        ctx[(rowBase + r) * HDP + k] = v;            // zero-pad k in [120,128)
    }
}

// ---------------- Kernel 2: WMMA scores + in-LDS masked softmax -------------
// grid = (S/16, B); block = 128 threads = 4 wave32.
__global__ void __launch_bounds__(128)
scores_softmax_kernel(const __bf16* __restrict__ ctx,
                      const int*    __restrict__ mask,
                      float*        __restrict__ out)
{
    extern __shared__ float sc[];            // [16][PITCH] fp32 score slab

    const int tid     = threadIdx.x;
    const int lane    = tid & 31;
    const int wv      = tid >> 5;            // wave 0..3
    const int batch   = blockIdx.y;
    const int rowTile = blockIdx.x;          // 0..255
    const int m16     = lane & 15;
    const int half    = lane >> 4;           // 0: lanes 0-15, 1: lanes 16-31

    // ---- A operand: 16 rows x 128 K, held in registers for the whole loop.
    // A 16x32, contiguous-K order used here (same as B): lanes 0-15 hold K [0..15],
    //                        lanes 16-31 hold K [16..31]  (M = lane&15)
    const __bf16* Abase =
        ctx + (long)(batch * S_ + rowTile * 16 + m16) * HDP;
    v16bf a0, a1, a2, a3;
    {
        v16bf* ap[4] = { &a0, &a1, &a2, &a3 };
#pragma unroll
        for (int c = 0; c < 4; ++c) {
            const v8bf lo = *(const v8bf*)(Abase + c * 32 + half * 16);
            const v8bf hi = *(const v8bf*)(Abase + c * 32 + half * 16 + 8);
#pragma unroll
            for (int e = 0; e < 8; ++e) { (*ap[c])[e] = lo[e]; (*ap[c])[8 + e] = hi[e]; }
        }
    }

    const __bf16* Bbatch = ctx + (long)batch * S_ * HDP;

    // B 32x16 layout: lane holds K contiguous [half*16 .. half*16+15] for
    // key N = lane&15 -> one 32-byte row-major load per chunk, no transpose.
    auto loadB = [&](int j, v16bf b[4]) {
        const __bf16* Brow = Bbatch + (long)(j * 16 + m16) * HDP;
#pragma unroll
        for (int c = 0; c < 4; ++c)
            b[c] = *(const v16bf*)(Brow + c * 32 + half * 16);
    };
    auto tile = [&](int j, const v16bf b[4]) {
        // speculative prefetch (OOB translation failures silently dropped)
        __builtin_prefetch(Bbatch + (long)((j + 12) * 16 + m16) * HDP, 0, 2);
        v8f acc = {};
        acc = __builtin_amdgcn_wmma_f32_16x16x32_bf16(false, a0, false, b[0],
                                                      (short)0, acc, false, false);
        acc = __builtin_amdgcn_wmma_f32_16x16x32_bf16(false, a1, false, b[1],
                                                      (short)0, acc, false, false);
        acc = __builtin_amdgcn_wmma_f32_16x16x32_bf16(false, a2, false, b[2],
                                                      (short)0, acc, false, false);
        acc = __builtin_amdgcn_wmma_f32_16x16x32_bf16(false, a3, false, b[3],
                                                      (short)0, acc, false, false);
        // C/D layout: VGPR r -> M = r + half*8, N = lane&15
        const int col = j * 16 + m16;
#pragma unroll
        for (int r = 0; r < 8; ++r)
            sc[(r + half * 8) * PITCH + col] = acc[r];
    };

    // ---- software-pipelined loop over 64 column tiles / wave (ping-pong B)
    {
        v16bf bA[4], bB[4];
        int j = wv;
        loadB(j, bA);
        for (int t = 0; t < 31; ++t) {        // 62 tiles in steady state
            loadB(j + 4, bB);                 // issue loads ahead of WMMAs
            tile(j, bA);
            loadB(j + 8, bA);
            tile(j + 4, bB);
            j += 8;
        }
        loadB(j + 4, bB);                     // last two tiles
        tile(j, bA);
        tile(j + 4, bB);
    }
    __syncthreads();

    // ---- masked softmax: 8 threads per row, float4/int4 vector traffic
    const int  row  = tid >> 3;              // 0..15
    const int  t8   = tid & 7;
    const long gRow = (long)batch * S_ + rowTile * 16 + row;
    const int4*   mrow4 = (const int4*)(mask + gRow * (long)S_);
    float4*       orow4 = (float4*)(out + gRow * (long)S_);
    float4*       srow4 = (float4*)(sc + row * PITCH);   // 16 B aligned

    float mx = -3.4e38f;
    for (int c = t8; c < S_ / 4; c += 8) {
        float4 s = srow4[c];                 // ds_load_b128
        const int4 m = mrow4[c];             // global_load_b128 (coalesced)
        s.x = (m.x == 0) ? -1e9f : s.x;      // masked_fill
        s.y = (m.y == 0) ? -1e9f : s.y;
        s.z = (m.z == 0) ? -1e9f : s.z;
        s.w = (m.w == 0) ? -1e9f : s.w;
        srow4[c] = s;                        // ds_store_b128
        mx = fmaxf(mx, fmaxf(fmaxf(s.x, s.y), fmaxf(s.z, s.w)));
    }
#pragma unroll
    for (int d = 1; d < 8; d <<= 1) mx = fmaxf(mx, __shfl_xor(mx, d, 32));

    float sum = 0.0f;
    for (int c = t8; c < S_ / 4; c += 8) {
        float4 s = srow4[c];
        s.x = __expf(s.x - mx);
        s.y = __expf(s.y - mx);
        s.z = __expf(s.z - mx);
        s.w = __expf(s.w - mx);
        srow4[c] = s;
        sum += (s.x + s.y) + (s.z + s.w);
    }
#pragma unroll
    for (int d = 1; d < 8; d <<= 1) sum += __shfl_xor(sum, d, 32);
    const float inv = 1.0f / sum;

    for (int c = t8; c < S_ / 4; c += 8) {
        float4 s = srow4[c];
        s.x *= inv; s.y *= inv; s.z *= inv; s.w *= inv;
        orow4[c] = s;                        // global_store_b128 (coalesced)
    }
}

// ---------------------------------------------------------------------------
extern "C" void kernel_launch(void* const* d_in, const int* in_sizes, int n_in,
                              void* d_out, int out_size, void* d_ws, size_t ws_size,
                              hipStream_t stream)
{
    const float* q  = (const float*)d_in[0];   // [B,S,512]
    const int*   mk = (const int*)  d_in[1];   // [B,S,S]
    const float* wq = (const float*)d_in[2];   // [120,512]
    const float* bq = (const float*)d_in[3];   // [120]
    const float* wt = (const float*)d_in[4];   // [1,120]
    float*   out = (float*)d_out;              // [B,S,S]
    __bf16*  ctx = (__bf16*)d_ws;              // [B*S, 128] bf16 = 8 MB

    // K1: 32768 rows / 8 rows per block
    proj_norm_kernel<<<(B_ * S_) / 8, 128, 0, stream>>>(q, wq, bq, wt, ctx);

    // K2: one workgroup per (16-row tile, batch); 257 KB dynamic LDS slab
    const size_t shmem = (size_t)16 * PITCH * sizeof(float);
    hipFuncSetAttribute((const void*)scores_softmax_kernel,
                        hipFuncAttributeMaxDynamicSharedMemorySize, (int)shmem);
    scores_softmax_kernel<<<dim3(S_ / 16, B_), 128, shmem, stream>>>(ctx, mk, out);
}
